// RelativeSelfAttention_34342558498894
// MI455X (gfx1250) — matrix operand order repaired and checked by reading, not compiled, not run on hardware
//
#include <hip/hip_runtime.h>
#include <hip/hip_bf16.h>

// ---------- types ----------
typedef __attribute__((ext_vector_type(16))) __bf16 v16bf;
typedef __attribute__((ext_vector_type(8)))  float  v8f;
typedef __attribute__((ext_vector_type(4)))  int    v4i;
typedef unsigned int   u32;
typedef unsigned short u16;

union BF16x16 { v16bf v; u16 s[16]; uint4 q[2]; };

static __device__ __forceinline__ u16 f2bf_bits(float f) {
    u32 u = __float_as_uint(f);
    u32 r = u + 0x7FFFu + ((u >> 16) & 1u);   // round-to-nearest-even
    return (u16)(r >> 16);
}

// 16 contiguous bf16 (32 bytes) -> one B-fragment lane slice (global or LDS)
static __device__ __forceinline__ v16bf load16bf(const u16* p) {
    BF16x16 t;
    t.q[0] = *(const uint4*)(p);
    t.q[1] = *(const uint4*)(p + 8);
    return t.v;
}

// A-fragment per-lane load: elems 0..15 = K hi*16..+15
static __device__ __forceinline__ v16bf load_a_frag(const u16* rowbase, int hi) {
    BF16x16 t;
    t.q[0] = *(const uint4*)(rowbase + hi * 16);
    t.q[1] = *(const uint4*)(rowbase + hi * 16 + 8);
    return t.v;
}

static __device__ __forceinline__ v8f wmma_bf16(v16bf a, v16bf b, v8f c) {
    return __builtin_amdgcn_wmma_f32_16x16x32_bf16(false, a, false, b, (short)0, c, false, false);
}

// ---------- async global->LDS staging (16B per call) ----------
#if defined(__has_builtin)
#if __has_builtin(__builtin_amdgcn_global_load_async_to_lds_b128)
#define HAVE_ASYNC_LDS 1
#endif
#endif
#ifndef HAVE_ASYNC_LDS
#define HAVE_ASYNC_LDS 0
#endif

typedef __attribute__((address_space(1))) v4i* gv4i_ptr;
typedef __attribute__((address_space(3))) v4i* lv4i_ptr;

static __device__ __forceinline__ void stage16B(u16* ldsp, const u16* gp) {
#if HAVE_ASYNC_LDS
    __builtin_amdgcn_global_load_async_to_lds_b128(
        (gv4i_ptr)(void*)gp, (lv4i_ptr)(void*)ldsp, 0, 0);
#else
    *(uint4*)ldsp = *(const uint4*)gp;
#endif
}

static __device__ __forceinline__ void stage_wait() {
#if HAVE_ASYNC_LDS
#if __has_builtin(__builtin_amdgcn_s_wait_asynccnt)
    __builtin_amdgcn_s_wait_asynccnt(0);
#else
    asm volatile("s_wait_asynccnt 0" ::: "memory");
#endif
#endif
}

// ---------- constants ----------
#define NB 4          // batch
#define NL 1024       // seq len
#define ND 512        // model dim
#define NH 8          // heads
#define DH 64         // head dim
#define ECOLS 64
#define EROWS 2048    // 2L-1 = 2047 rows + 1 zero pad row
#define LOG2E 1.4426950408889634f

// ---------- prep kernels ----------
__global__ void cvt_x_kernel(const float* __restrict__ x, u16* __restrict__ xb, int n) {
    int i = blockIdx.x * 256 + threadIdx.x;
    if (i < n) xb[i] = f2bf_bits(x[i]);
}

__global__ void cvt_w_kernel(const float* __restrict__ W, u16* __restrict__ wt) {
    int i = blockIdx.x * 256 + threadIdx.x;
    if (i >= 3 * ND * ND) return;            // 1536*512
    int c = i >> 9;                           // output column 0..1535
    int k = i & 511;                          // input dim
    wt[i] = f2bf_bits(W[(size_t)k * (3 * ND) + c]);
}

__global__ void emb_kernel(u16* __restrict__ E) {
    int i = blockIdx.x * 256 + threadIdx.x;
    if (i >= EROWS * ECOLS) return;
    int row = i >> 6;
    int d   = i & 63;
    float val = 0.0f;
    if (row < 2 * NL - 1) {
        float pos = (float)(row - (NL - 1));
        int j = (d < 32) ? d : (d - 32);
        float div = expf(-(float)(2 * j) * (9.210340371976184f / 64.0f)); // ln(10000)/64
        float a = pos * div;
        val = (d < 32) ? sinf(a) : cosf(a);
    }
    E[i] = f2bf_bits(val);
}

// ---------- projection GEMM: proj = x @ W + b, scattered to q/k/vT ----------
// per wave: 16 rows (M) x 64 cols (N); K = 512 in steps of 32 (bf16 WMMA)
__global__ __launch_bounds__(256) void proj_kernel(const u16* __restrict__ xb,
                                                   const u16* __restrict__ wt,
                                                   const float* __restrict__ bias,
                                                   u16* __restrict__ qb,
                                                   u16* __restrict__ kb,
                                                   u16* __restrict__ vT) {
    const int lane = threadIdx.x;
    const int wid  = blockIdx.x * 8 + threadIdx.y;   // 0..6143
    const int mt = wid / 24;                          // row tile 0..255  (BL/16)
    const int ng = wid % 24;                          // 64-col group 0..23
    const int hi = lane >> 4;
    const int ln = lane & 15;
    const int nb = ng * 64;

    v8f acc[4];
#pragma unroll
    for (int t = 0; t < 4; t++) acc[t] = (v8f){};

    const u16* arow = xb + (size_t)(mt * 16 + ln) * ND;
#pragma unroll 4
    for (int k0 = 0; k0 < ND; k0 += 32) {
        v16bf a = load_a_frag(arow + k0, hi);
#pragma unroll
        for (int t = 0; t < 4; t++) {
            v16bf b = load16bf(wt + (size_t)(nb + t * 16 + ln) * ND + k0 + hi * 16);
            acc[t] = wmma_bf16(a, b, acc[t]);
        }
    }

    // epilogue: bias add, bf16 convert, scatter to q/k/vT layouts
    const int h    = ng / 3;
    const int part = ng % 3;      // 0=q 1=k 2=v
#pragma unroll
    for (int t = 0; t < 4; t++) {
        int d = t * 16 + ln;                 // 0..63 within head
        float bv = bias[nb + d];
#pragma unroll
        for (int r = 0; r < 8; r++) {
            int row  = hi * 8 + r;
            int gRow = mt * 16 + row;        // 0..4095
            int bi = gRow >> 10;
            int l  = gRow & (NL - 1);
            u16 val = f2bf_bits(acc[t][r] + bv);
            if (part == 0)
                qb[(((size_t)bi * NH + h) * NL + l) * DH + d] = val;
            else if (part == 1)
                kb[(((size_t)bi * NH + h) * NL + l) * DH + d] = val;
            else
                vT[(((size_t)bi * NH + h) * DH + d) * NL + l] = val;
        }
    }
}

// ---------- fused attention with relative positions (flash-style) ----------
// one wave = one 16-row q-tile; 8 waves/block share one (b,h) and cooperatively
// stage each 32-wide K/V chunk into LDS (async global->LDS when available).
__global__ __launch_bounds__(256) void attn_kernel(const u16* __restrict__ qb,
                                                   const u16* __restrict__ kb,
                                                   const u16* __restrict__ vT,
                                                   const u16* __restrict__ E,
                                                   float* __restrict__ out) {
    __shared__ __align__(16) u16 Kbuf[32 * DH];     // 4 KB : K chunk [32 k][64 d]
    __shared__ __align__(16) u16 Vbuf[DH * 32];     // 4 KB : V chunk [64 d][32 k]
    __shared__ float Pbuf[8][16 * 48];              // 24 KB: per-wave scratch

    const int lane = threadIdx.x;
    const int w    = threadIdx.y;
    const int wid  = blockIdx.x * 8 + w;     // 0..2047
    const int b  = wid >> 9;                 // /(H*64)
    const int h  = (wid >> 6) & 7;           // same for all 8 waves of a block
    const int q0 = (wid & 63) * 16;
    const int hi = lane >> 4;
    const int ln = lane & 15;
    const int seg = w * 32 + lane;           // 0..255 staging segment id

    const u16* qbase = qb + (((size_t)b * NH + h) * NL + q0) * DH;
    const u16* kbase = kb + (((size_t)b * NH + h) * NL) * DH;
    const u16* vbase = vT + (((size_t)b * NH + h) * DH) * NL;
    float* P = Pbuf[w];

    // Q A-fragments, loaded once (d 0..31 and 32..63)
    const u16* qrow = qbase + (size_t)ln * DH;
    v16bf qa0 = load_a_frag(qrow + 0, hi);
    v16bf qa1 = load_a_frag(qrow + 32, hi);

    v8f o0 = (v8f){}, o1 = (v8f){}, o2 = (v8f){}, o3 = (v8f){};
    float mrow[8], lrow[8];
#pragma unroll
    for (int r = 0; r < 8; r++) { mrow[r] = -1e30f; lrow[r] = 0.0f; }

    const float scale = 0.125f;              // 1/sqrt(64)

    for (int k0 = 0; k0 < NL; k0 += 32) {
        // all waves finished reading the previous chunk before overwrite
        __syncthreads();

        // ---- cooperative staging of K[32x64] and V[64x32] chunks (16B/thread each)
        stage16B(Kbuf + seg * 8, kbase + (size_t)k0 * DH + seg * 8);
        stage16B(Vbuf + (seg >> 2) * 32 + (seg & 3) * 8,
                 vbase + (size_t)(seg >> 2) * NL + k0 + (seg & 3) * 8);

        // ---- overlap async copy with the relative-position band:
        // P_rel[16 x 48], col t <- E row (q0-k0+992+t)
        const int rbase = q0 - k0 + 992;
#pragma unroll
        for (int tt = 0; tt < 3; tt++) {
            const u16* erow = E + (size_t)(rbase + tt * 16 + ln) * ECOLS;
            v16bf eb0 = load16bf(erow + hi * 16);
            v16bf eb1 = load16bf(erow + 32 + hi * 16);
            if (k0 + 32 < NL)
                __builtin_prefetch(erow - 32 * ECOLS, 0, 3);  // next chunk's band
            v8f pr = (v8f){};
            pr = wmma_bf16(qa0, eb0, pr);
            pr = wmma_bf16(qa1, eb1, pr);
#pragma unroll
            for (int r = 0; r < 8; r++)
                P[(hi * 8 + r) * 48 + tt * 16 + ln] = pr[r];
        }

        stage_wait();
        __syncthreads();                     // K/V chunk visible to all waves

        // ---- S = Q K^T for two 16-wide tiles (K fragments from LDS)
        v8f s0 = (v8f){}, s1 = (v8f){};
        {
            const u16* kr0 = Kbuf + (size_t)ln * DH;
            s0 = wmma_bf16(qa0, load16bf(kr0 + hi * 16), s0);
            s0 = wmma_bf16(qa1, load16bf(kr0 + 32 + hi * 16), s0);
            const u16* kr1 = Kbuf + (size_t)(16 + ln) * DH;
            s1 = wmma_bf16(qa0, load16bf(kr1 + hi * 16), s1);
            s1 = wmma_bf16(qa1, load16bf(kr1 + 32 + hi * 16), s1);
        }

        // ---- combine with rel (Toeplitz skew gather from LDS), scale
        float sv0[8], sv1[8];
#pragma unroll
        for (int r = 0; r < 8; r++) {
            int row = hi * 8 + r;
            sv0[r] = (s0[r] + P[row * 48 + (31 + row - ln)]) * scale;  // j = ln
            sv1[r] = (s1[r] + P[row * 48 + (15 + row - ln)]) * scale;  // j = 16+ln
        }

        // ---- online softmax update (row = reduction across 16 lanes of half)
#pragma unroll
        for (int r = 0; r < 8; r++) {
            float mx = fmaxf(sv0[r], sv1[r]);
#pragma unroll
            for (int off = 1; off < 16; off <<= 1)
                mx = fmaxf(mx, __shfl_xor(mx, off, 32));
            float mnew  = fmaxf(mrow[r], mx);
            float alpha = exp2f((mrow[r] - mnew) * LOG2E);
            float p0 = exp2f((sv0[r] - mnew) * LOG2E);
            float p1 = exp2f((sv1[r] - mnew) * LOG2E);
            float ls = p0 + p1;
#pragma unroll
            for (int off = 1; off < 16; off <<= 1)
                ls += __shfl_xor(ls, off, 32);
            lrow[r] = lrow[r] * alpha + ls;
            mrow[r] = mnew;
            o0[r] *= alpha; o1[r] *= alpha; o2[r] *= alpha; o3[r] *= alpha;
            sv0[r] = p0; sv1[r] = p1;
        }

        // ---- stage probabilities [16 x 32] in per-wave LDS, re-read as A-fragment
#pragma unroll
        for (int r = 0; r < 8; r++) {
            int row = hi * 8 + r;
            P[row * 32 + ln]      = sv0[r];
            P[row * 32 + 16 + ln] = sv1[r];
        }
        BF16x16 pf;
#pragma unroll
        for (int e = 0; e < 16; e++) {
            int K = hi * 16 + e;
            pf.s[e] = f2bf_bits(P[ln * 32 + K]);
        }

        // ---- O += P @ V (4 d-tiles, K = 32, V fragments from LDS)
        {
            v16bf vb0 = load16bf(Vbuf + (size_t)(0  + ln) * 32 + hi * 16);
            v16bf vb1 = load16bf(Vbuf + (size_t)(16 + ln) * 32 + hi * 16);
            v16bf vb2 = load16bf(Vbuf + (size_t)(32 + ln) * 32 + hi * 16);
            v16bf vb3 = load16bf(Vbuf + (size_t)(48 + ln) * 32 + hi * 16);
            o0 = wmma_bf16(pf.v, vb0, o0);
            o1 = wmma_bf16(pf.v, vb1, o1);
            o2 = wmma_bf16(pf.v, vb2, o2);
            o3 = wmma_bf16(pf.v, vb3, o3);
        }
    }

    // ---- epilogue: normalize, store f32 out[b][l][h*64+d]
    float* obase = out + ((size_t)b * NL + q0) * ND + h * DH;
#pragma unroll
    for (int r = 0; r < 8; r++) {
        int row = hi * 8 + r;
        float inv = 1.0f / lrow[r];
        obase[(size_t)row * ND + 0  + ln] = o0[r] * inv;
        obase[(size_t)row * ND + 16 + ln] = o1[r] * inv;
        obase[(size_t)row * ND + 32 + ln] = o2[r] * inv;
        obase[(size_t)row * ND + 48 + ln] = o3[r] * inv;
    }
}

// ---------- launcher ----------
extern "C" void kernel_launch(void* const* d_in, const int* in_sizes, int n_in,
                              void* d_out, int out_size, void* d_ws, size_t ws_size,
                              hipStream_t stream) {
    (void)in_sizes; (void)n_in; (void)out_size; (void)ws_size;
    const float* x    = (const float*)d_in[0];
    const float* W    = (const float*)d_in[1];
    const float* bias = (const float*)d_in[2];
    float* out = (float*)d_out;

    char* ws = (char*)d_ws;
    // workspace layout (bytes), all offsets 256-aligned
    u16* xb = (u16*)(ws + 0);                 //  4 MB : x in bf16 [4096][512]
    u16* wt = (u16*)(ws + 4194304);           //  1.5MB: W^T bf16 [1536][512]
    u16* E  = (u16*)(ws + 5767168);           //  256KB: emb bf16 [2048][64] (row 2047 = 0)
    u16* qb = (u16*)(ws + 6029312);           //  4 MB : q bf16 [B][H][L][64]
    u16* kb = (u16*)(ws + 10223616);          //  4 MB : k bf16 [B][H][L][64]
    u16* vT = (u16*)(ws + 14417920);          //  4 MB : v bf16 [B][H][64][L]

    cvt_x_kernel<<<(NB * NL * ND + 255) / 256, 256, 0, stream>>>(x, xb, NB * NL * ND);
    cvt_w_kernel<<<(3 * ND * ND + 255) / 256, 256, 0, stream>>>(W, wt);
    emb_kernel<<<(EROWS * ECOLS + 255) / 256, 256, 0, stream>>>(E);
    proj_kernel<<<768, dim3(32, 8), 0, stream>>>(xb, wt, bias, qb, kb, vT);
    attn_kernel<<<256, dim3(32, 8), 0, stream>>>(qb, kb, vT, E, out);
}
